// GINE_63660005261516
// MI455X (gfx1250) — matrix operand order repaired and checked by reading, not compiled, not run on hardware
//
#include <hip/hip_runtime.h>

// ---------------------------------------------------------------------------
// GINE GNN for MI455X (gfx1250, wave32).
//   - weights pre-packed ONCE to bf16-pair (u32) layout in global memory
//   - per-layer weight staging via TDM (tensor_load_to_lds), double-buffered
//     in LDS and overlapped with compute; waits via s_wait_tensorcnt
//   - fused 3-layer MLPs via v_wmma_f32_16x16x32_bf16 (f32 accumulate);
//     K-loop = 2x ds_load_b128 (A) + paired ds loads (B) + v_wmma, branch-free
//   - hidden tiles re-packed via __shfl_xor lane pairing (wave32)
//   - edge message + segment-sum via fp32 global atomics (agg fits in 192MB L2)
// ---------------------------------------------------------------------------

typedef __attribute__((ext_vector_type(16))) __bf16 v16bf;
typedef __attribute__((ext_vector_type(8)))  float  v8f;
typedef unsigned int u32x4 __attribute__((ext_vector_type(4)));
typedef int          i32x8 __attribute__((ext_vector_type(8)));
typedef int          i32x4 __attribute__((ext_vector_type(4)));

static __device__ __forceinline__ unsigned f2bf1(float f) {
  // round-to-nearest-even f32 -> bf16 (in low 16 bits)
  unsigned u = __float_as_uint(f);
  u += 0x7FFFu + ((u >> 16) & 1u);
  return u >> 16;
}
static __device__ __forceinline__ unsigned packbf(float lo, float hi) {
  return f2bf1(lo) | (f2bf1(hi) << 16);
}

union BFrag {
  v16bf    v;
  unsigned d[8];
  uint4    q[2];
};

// TDM: DMA a [rows][128] u32 tile from global into LDS (2-D descriptor).
// D# layout per CDNA5 ISA ch.8 (group0: count/lds/global/type, group1: dims).
// This toolchain exposes the 6-arg builtin: (g0, g1, g2, g3, g4, cpol).
static __device__ __forceinline__ void tdm_load_w(const unsigned* __restrict__ gsrc,
                                                  unsigned* lds_dst, int rows) {
  const unsigned long long ga = (unsigned long long)(size_t)gsrc;
  u32x4 g0;
  g0[0] = 1u;                                       // count=1 (valid user D#)
  g0[1] = (unsigned)(size_t)(void*)lds_dst;         // lds_addr (bytes)
  g0[2] = (unsigned)ga;                             // global_addr[31:0]
  g0[3] = (unsigned)((ga >> 32) & 0x01FFFFFFu)      // global_addr[56:32]
        | (2u << 30);                               // type=2 ("image")
  i32x8 g1;
  g1[0] = (int)(2u << 16);                          // wg_mask=0, data_size=4B
  g1[1] = (int)(128u << 16);                        // tensor_dim0 = 128
  g1[2] = (int)(((unsigned)rows) << 16);            // tensor_dim1 = rows
  g1[3] = (int)(128u << 16);                        // tile_dim0 = 128
  g1[4] = rows;                                     // tile_dim1 = rows
  g1[5] = 128;                                      // tensor_dim0_stride = 128
  g1[6] = 0;
  g1[7] = 0;
  const i32x4 z4 = {};
  const i32x8 z8 = {};
  __builtin_amdgcn_tensor_load_to_lds(g0, g1, z4, z4, z8, 0);
}

// A-fragment: two 16B LDS loads from packed tile [16][64] (k-pairs fast axis).
#define LOAD_A(frag, tileSrc, kkp)                                           \
  {                                                                          \
    const int offA = (kkp) + hi * 8;                                         \
    (frag).q[0] = *(const uint4*)&(tileSrc)[r16 * 64 + offA];                \
    (frag).q[1] = *(const uint4*)&(tileSrc)[r16 * 64 + offA + 4];            \
  }

// B-fragment: 8 dword LDS loads from packed weights [64][128].
#define LOAD_B(frag, wsrc, kkp)                                              \
  {                                                                          \
    const int offB = (kkp) + hi * 8;                                         \
    _Pragma("unroll")                                                        \
    for (int v = 0; v < 8; ++v)                                              \
      (frag).d[v] = (wsrc)[(offB + v) * 128 + col];                          \
  }

// One block computes a 16-row x 128-col tile through all three MLP layers.
// 8 waves; wave w owns output columns [16w, 16w+16).
__global__ __launch_bounds__(256) void fused_mlp3_wmma(
    const float* __restrict__ in,       // [R, Kin]
    const float* __restrict__ addp,     // optional addend [R, Kin] (GINE h+agg)
    const unsigned* __restrict__ W1p,   // packed [64][128] u32 slot
    const float* __restrict__ b1,
    const unsigned* __restrict__ W2p, const float* __restrict__ b2,
    const unsigned* __restrict__ W3p, const float* __restrict__ b3,
    float* __restrict__ out,            // [R, Nout]
    int Kin, int Nout)
{
  __shared__ __align__(16) unsigned wbuf0[64 * 128]; // packed weights (32KB)
  __shared__ __align__(16) unsigned wbuf1[64 * 128]; // double buffer   (32KB)
  __shared__ __align__(16) unsigned tA[16 * 64];     // packed act tile (4KB)
  __shared__ __align__(16) unsigned tB[16 * 64];     // ping-pong tile  (4KB)

  const int tid  = threadIdx.x;
  const int lane = tid & 31;
  const int wv   = tid >> 5;       // 0..7 -> N-tile
  const int r16  = lane & 15;
  const int hi   = lane >> 4;      // lane group (0 or 1)
  const int col  = wv * 16 + r16;
  const long long row0 = (long long)blockIdx.x * 16;

  const int Kpad1  = (Kin + 31) & ~31;
  const int kkmax1 = Kpad1 >> 5;   // 1 (Kin<=32) or 4 (Kin=128)

  // ---- kick off W1 and W2 DMAs (wave 0; TDM ignores EXEC, per-wave counter) --
  if (wv == 0 && lane == 0) {
    tdm_load_w(W1p, wbuf0, Kpad1 >> 1);  // TENSORcnt -> 1
    tdm_load_w(W2p, wbuf1, 64);          // TENSORcnt -> 2
  }

  // ---- stage input tile packed [16][64], zero beyond Kin (uniform addp split) --
  if (addp) {
    for (int i = tid; i < 16 * 64; i += 256) {
      const int r = i >> 6, k0 = (i & 63) << 1;
      float x0 = 0.f, x1 = 0.f;
      if (k0 < Kin) { // Kin is even -> pair fully in-bounds
        const long long base = (row0 + r) * (long long)Kin + k0;
        const float2 t = *(const float2*)(in + base);
        const float2 s = *(const float2*)(addp + base);
        x0 = t.x + s.x; x1 = t.y + s.y;
      }
      tA[i] = packbf(x0, x1);
    }
  } else {
    for (int i = tid; i < 16 * 64; i += 256) {
      const int r = i >> 6, k0 = (i & 63) << 1;
      float x0 = 0.f, x1 = 0.f;
      if (k0 < Kin) {
        const float2 t = *(const float2*)(in + (row0 + r) * (long long)Kin + k0);
        x0 = t.x; x1 = t.y;
      }
      tA[i] = packbf(x0, x1);
    }
  }
  if (wv == 0) __builtin_amdgcn_s_wait_tensorcnt(1); // W1 landed
  __syncthreads();

  // ---- layer 1: tA x wbuf0 -> relu -> tB (packed via lane pairing) ----
  {
    v8f acc = {};
    for (int kk = 0; kk < kkmax1; ++kk) {
      BFrag a, b;
      LOAD_A(a, tA, kk * 16)
      LOAD_B(b, wbuf0, kk * 16)
      acc = __builtin_amdgcn_wmma_f32_16x16x32_bf16(false, a.v, false, b.v,
                                                    (short)0, acc, false, false);
    }
    const float bias = b1[col];
#pragma unroll
    for (int v = 0; v < 8; ++v) {
      const float val   = fmaxf(acc[v] + bias, 0.f);
      const float other = __shfl_xor(val, 1, 32); // partner column (col^1)
      if ((lane & 1) == 0)
        tB[(v + hi * 8) * 64 + (col >> 1)] = packbf(val, other);
    }
  }
  if (wv == 0) __builtin_amdgcn_s_wait_tensorcnt(0); // W2 landed
  __syncthreads();
  // wbuf0 reads all happened before this barrier -> safe to overwrite with W3,
  // overlapping the DMA with layer-2 compute.
  if (wv == 0 && lane == 0) tdm_load_w(W3p, wbuf0, 64);

  // ---- layer 2: tB x wbuf1 -> relu -> tA ----
  {
    v8f acc = {};
#pragma unroll
    for (int kk = 0; kk < 4; ++kk) {
      BFrag a, b;
      LOAD_A(a, tB, kk * 16)
      LOAD_B(b, wbuf1, kk * 16)
      acc = __builtin_amdgcn_wmma_f32_16x16x32_bf16(false, a.v, false, b.v,
                                                    (short)0, acc, false, false);
    }
    const float bias = b2[col];
#pragma unroll
    for (int v = 0; v < 8; ++v) {
      const float val   = fmaxf(acc[v] + bias, 0.f);
      const float other = __shfl_xor(val, 1, 32);
      if ((lane & 1) == 0)
        tA[(v + hi * 8) * 64 + (col >> 1)] = packbf(val, other);
    }
  }
  if (wv == 0) __builtin_amdgcn_s_wait_tensorcnt(0); // W3 landed
  __syncthreads();

  // ---- layer 3: tA x wbuf0 -> global out (no relu) ----
  if (wv * 16 < Nout) { // wave-uniform: EXEC stays all-1s around WMMA
    v8f acc = {};
#pragma unroll
    for (int kk = 0; kk < 4; ++kk) {
      BFrag a, b;
      LOAD_A(a, tA, kk * 16)
      LOAD_B(b, wbuf0, kk * 16)
      acc = __builtin_amdgcn_wmma_f32_16x16x32_bf16(false, a.v, false, b.v,
                                                    (short)0, acc, false, false);
    }
    const float bias = b3[col];
#pragma unroll
    for (int v = 0; v < 8; ++v)
      out[(row0 + v + hi * 8) * (long long)Nout + col] = acc[v] + bias;
  }
}

// One-time: pack W [Kin, Nout] f32 -> [64][128] u32 slot of bf16 pairs
// (k pairs packed, zero-padded in both K and N).
__global__ __launch_bounds__(256) void pack_weights(
    const float* __restrict__ W, unsigned* __restrict__ dst, int Kin, int Nout)
{
  const int i = blockIdx.x * 256 + threadIdx.x;   // grid covers 64*128
  if (i >= 64 * 128) return;
  const int kp = i >> 7, n = i & 127;
  const int k0 = kp << 1;
  const float w0 = (k0 < Kin && n < Nout) ? W[k0 * Nout + n] : 0.f;
  const float w1 = (k0 + 1 < Kin && n < Nout) ? W[(k0 + 1) * Nout + n] : 0.f;
  dst[i] = packbf(w0, w1);
}

// msg = relu(h[src] + e); agg[dst] += msg.  One edge per 32 lanes, 4 floats/lane.
__global__ __launch_bounds__(256) void edge_scatter(
    const float* __restrict__ h, const float* __restrict__ e,
    const int* __restrict__ src, const int* __restrict__ dst,
    float* __restrict__ agg, long long E)
{
  const long long t = (long long)blockIdx.x * 256 + threadIdx.x;
  const long long edge = t >> 5;
  const int lane = (int)(t & 31);
  if (edge >= E) return;
  const int s = src[edge];
  const int d = dst[edge];
  const float4 hv = *(const float4*)(h + (long long)s * 128 + lane * 4);
  const float4 ev = *(const float4*)(e + edge * 128 + lane * 4);
  float4 m;
  m.x = fmaxf(hv.x + ev.x, 0.f);
  m.y = fmaxf(hv.y + ev.y, 0.f);
  m.z = fmaxf(hv.z + ev.z, 0.f);
  m.w = fmaxf(hv.w + ev.w, 0.f);
  float* ap = agg + (long long)d * 128 + lane * 4;
  atomicAdd(ap + 0, m.x);
  atomicAdd(ap + 1, m.y);
  atomicAdd(ap + 2, m.z);
  atomicAdd(ap + 3, m.w);
}

__global__ __launch_bounds__(256) void zero4(float4* __restrict__ p, long long n4) {
  const long long i = (long long)blockIdx.x * 256 + threadIdx.x;
  if (i < n4) p[i] = make_float4(0.f, 0.f, 0.f, 0.f);
}

extern "C" void kernel_launch(void* const* d_in, const int* in_sizes, int n_in,
                              void* d_out, int out_size, void* d_ws, size_t ws_size,
                              hipStream_t stream) {
  (void)in_sizes; (void)n_in; (void)out_size; (void)ws_size;
  const long long NN = 100000, NE = 1600000;

  const float* x   = (const float*)d_in[0];
  const float* ef  = (const float*)d_in[1];
  const int*   src = (const int*)d_in[2];
  const int*   dst = (const int*)d_in[3];
  const float* Wf[9]  = { (const float*)d_in[4],  (const float*)d_in[6],  (const float*)d_in[8],
                          (const float*)d_in[10], (const float*)d_in[12], (const float*)d_in[14],
                          (const float*)d_in[22], (const float*)d_in[24], (const float*)d_in[26] };
  const float* n2nb1 = (const float*)d_in[5];
  const float* n2nb2 = (const float*)d_in[7];
  const float* n2nb3 = (const float*)d_in[9];
  const float* e2nb1 = (const float*)d_in[11];
  const float* e2nb2 = (const float*)d_in[13];
  const float* e2nb3 = (const float*)d_in[15];
  const float* gW1 = (const float*)d_in[16]; const float* gb1 = (const float*)d_in[17];
  const float* gW2 = (const float*)d_in[18]; const float* gb2 = (const float*)d_in[19];
  const float* gW3 = (const float*)d_in[20]; const float* gb3 = (const float*)d_in[21];
  const float* fb1 = (const float*)d_in[23];
  const float* fb2 = (const float*)d_in[25];
  const float* fb3 = (const float*)d_in[27];

  // ---- workspace layout ----
  const size_t SLOT = 64 * 128;              // u32 per packed weight slot (32KB)
  unsigned* wslab = (unsigned*)d_ws;         // 18 slots = 576KB
  const long long NH = NN * 128;             // 12.8M floats (51.2 MB)
  float* h   = (float*)(wslab + 18 * SLOT);  // [NN,128]
  float* h2  = h   + NH;                     // [NN,128]
  float* agg = h2  + NH;                     // [NN,128]  (L2-resident)
  float* e   = agg + NH;                     // [NE,128]  (819 MB, streamed)

  // slots: 0-2 n2n, 3-5 e2n, 6-14 gine (3 layers x 3), 15-17 fin
  unsigned* S[18];
  for (int i = 0; i < 18; ++i) S[i] = wslab + (size_t)i * SLOT;

  // ---- pack all weights once (bf16 pairs, padded slots) ----
  pack_weights<<<32, 256, 0, stream>>>(Wf[0], S[0], 16, 128);
  pack_weights<<<32, 256, 0, stream>>>(Wf[1], S[1], 128, 128);
  pack_weights<<<32, 256, 0, stream>>>(Wf[2], S[2], 128, 128);
  pack_weights<<<32, 256, 0, stream>>>(Wf[3], S[3], 8, 128);
  pack_weights<<<32, 256, 0, stream>>>(Wf[4], S[4], 128, 128);
  pack_weights<<<32, 256, 0, stream>>>(Wf[5], S[5], 128, 128);
  for (int l = 0; l < 3; ++l) {
    pack_weights<<<32, 256, 0, stream>>>(gW1 + (long long)l * 128 * 128, S[6 + l * 3 + 0], 128, 128);
    pack_weights<<<32, 256, 0, stream>>>(gW2 + (long long)l * 128 * 128, S[6 + l * 3 + 1], 128, 128);
    pack_weights<<<32, 256, 0, stream>>>(gW3 + (long long)l * 128 * 128, S[6 + l * 3 + 2], 128, 128);
  }
  pack_weights<<<32, 256, 0, stream>>>(Wf[6], S[15], 128, 128);
  pack_weights<<<32, 256, 0, stream>>>(Wf[7], S[16], 128, 128);
  pack_weights<<<32, 256, 0, stream>>>(Wf[8], S[17], 128, 32);

  // ---- input projections ----
  fused_mlp3_wmma<<<(int)(NN / 16), 256, 0, stream>>>(
      x, nullptr, S[0], n2nb1, S[1], n2nb2, S[2], n2nb3, h, 16, 128);
  fused_mlp3_wmma<<<(int)(NE / 16), 256, 0, stream>>>(
      ef, nullptr, S[3], e2nb1, S[4], e2nb2, S[5], e2nb3, e, 8, 128);

  // ---- 3 GINE layers ----
  const long long n4 = NH / 4;
  const int zgrid = (int)((n4 + 255) / 256);
  const int sgrid = (int)((NE * 32 + 255) / 256);
  for (int l = 0; l < 3; ++l) {
    zero4<<<zgrid, 256, 0, stream>>>((float4*)agg, n4);
    edge_scatter<<<sgrid, 256, 0, stream>>>(h, e, src, dst, agg, NE);
    fused_mlp3_wmma<<<(int)(NN / 16), 256, 0, stream>>>(
        h, agg,
        S[6 + l * 3 + 0], gb1 + (long long)l * 128,
        S[6 + l * 3 + 1], gb2 + (long long)l * 128,
        S[6 + l * 3 + 2], gb3 + (long long)l * 128,
        h2, 128, 128);
    float* t = h; h = h2; h2 = t;
  }

  // ---- final projection -> d_out [NN, 32] float ----
  fused_mlp3_wmma<<<(int)(NN / 16), 256, 0, stream>>>(
      h, nullptr, S[15], fb1, S[16], fb2, S[17], fb3, (float*)d_out, 128, 32);
}
